// SchNet_45707041964613
// MI455X (gfx1250) — matrix operand order repaired and checked by reading, not compiled, not run on hardware
//
#include <hip/hip_runtime.h>
#include <math.h>

#define NBATCH 32
#define NATOM  128
#define FDIM   64
#define KRBF   50
#define EPSC   1e-6f

typedef _Float16 half16 __attribute__((ext_vector_type(16)));
typedef _Float16 half8v __attribute__((ext_vector_type(8)));
typedef float    float8 __attribute__((ext_vector_type(8)));

union H16 { half16 v; _Float16 e[16]; half8v h[2]; };
union F8  { float8 v; float e[8]; };

// softplus with hardware exp/log (v_exp_f32 / v_log_f32); values are O(1),
// f16 WMMA inputs dominate the error budget so native-accuracy is plenty.
__device__ __forceinline__ float softplusf_(float z) {
  float s = __logf(1.f + __expf(z));
  return (z > 15.f) ? z : s;
}

__device__ __forceinline__ float sigmoidf_(float z) {
  return __builtin_amdgcn_rcpf(1.f + __expf(-z));
}

__device__ __forceinline__ _Float16 rbf16(float dm, int k, bool zr, float gam,
                                          const float* mu) {
  if (zr || k >= KRBF) return (_Float16)0.f;
  float t = dm - mu[k];
  return (_Float16)__expf(-gam * t * t);
}

// Build A-fragments (16x32 f16, contiguous k = 16*lh + s order) for the rbf matrix rows of one
// M-tile. lh = lane>>4 selects the 16-wide K-half (k = 16*lh + s), same order as the B loads.
__device__ __forceinline__ void build_rbf_frags(H16& a0, H16& a1, float dm,
                                                bool zr, int lh, float gam,
                                                const float* mu) {
#pragma unroll
  for (int p = 0; p < 8; ++p) {
    int k0 = lh * 16 + p;
    a0.e[p]     = rbf16(dm, k0,      zr, gam, mu);
    a0.e[p + 8] = rbf16(dm, k0 + 8,  zr, gam, mu);
    a1.e[p]     = rbf16(dm, k0 + 32, zr, gam, mu);
    a1.e[p + 8] = rbf16(dm, k0 + 40, zr, gam, mu);
  }
}

// Kernel 1: one workgroup (128 thr = 4 wave32) per (batch, atom).
// Computes e[b][i][j] = dE/dd_ij for every neighbor j, into E (d_ws scratch).
__global__ __launch_bounds__(128) void schnet_pair_grad(
    const float* __restrict__ xs, const float* __restrict__ mus,
    const float* __restrict__ gamma_p,
    const float* __restrict__ Wenc, const float* __restrict__ benc,
    const float* __restrict__ cw1_0, const float* __restrict__ cb1_0,
    const float* __restrict__ cw2_0, const float* __restrict__ cb2_0,
    const float* __restrict__ cw1_1, const float* __restrict__ cb1_1,
    const float* __restrict__ cw2_1, const float* __restrict__ cb2_1,
    const float* __restrict__ cw1_2, const float* __restrict__ cb1_2,
    const float* __restrict__ cw2_2, const float* __restrict__ cb2_2,
    const float* __restrict__ Wt1, const float* __restrict__ bt1,
    const float* __restrict__ Wt2,
    float* __restrict__ E)
{
  // LDS: ~62 KB total (within the 64 KB static limit; WGP has 320 KB).
  __shared__ __align__(32) _Float16 w1t[3][FDIM][FDIM];   // W1 transposed [f][k], fwd B
  __shared__ __align__(32) _Float16 w1o[3][FDIM][FDIM];   // W1 original  [k][f], bwd B
  __shared__ __align__(32) _Float16 dzbuf[4][16][FDIM];   // per-wave dz tile
  __shared__ float d_sh[NATOM];
  __shared__ float mu_sh[FDIM];
  __shared__ float b1sh[3][FDIM];
  __shared__ float Ssh[3][FDIM];     // sum_j softplus(z) per feature
  __shared__ float Aar[3][FDIM];     // 1 + Wsum^l
  __shared__ float dWsh[3][FDIM];    // dE/dWsum^l
  __shared__ float dsvsh[3][FDIM];   // dE/ds per feature (same for all j)
  __shared__ float h3sh[FDIM];
  __shared__ float goutsh[FDIM];
  __shared__ float e_sh[NATOM];
  __shared__ float xi_sh[3];

  const int tid = threadIdx.x;
  const int b = blockIdx.x >> 7;
  const int i = blockIdx.x & (NATOM - 1);
  const float gam = gamma_p[0];

  const float* W1s[3] = {cw1_0, cw1_1, cw1_2};
  const float* B1s[3] = {cb1_0, cb1_1, cb1_2};
  const float* W2s[3] = {cw2_0, cw2_1, cw2_2};
  const float* B2s[3] = {cb2_0, cb2_1, cb2_2};

  // ---- stage inputs into LDS ----
  if (tid < 3)    xi_sh[tid] = xs[(b * NATOM + i) * 3 + tid];
  if (tid < FDIM) mu_sh[tid] = (tid < KRBF) ? mus[tid] : 0.f;
  e_sh[tid] = 0.f;
  for (int t = tid; t < 3 * FDIM; t += 128) {
    b1sh[t >> 6][t & 63] = B1s[t >> 6][t & 63];
    Ssh[t >> 6][t & 63] = 0.f;
  }
#pragma unroll
  for (int l = 0; l < 3; ++l)
    for (int t = tid; t < FDIM * FDIM; t += 128) {
      int kr = t >> 6, f = t & 63;
      float v = (kr < KRBF) ? W1s[l][kr * FDIM + f] : 0.f;
      w1o[l][kr][f] = (_Float16)v;
      w1t[l][f][kr] = (_Float16)v;
    }
  __syncthreads();

  // distances: slot jj -> neighbor j=(i+1+jj)%N; slot 127 is the padded self-pair
  {
    int j = (i + 1 + tid) & (NATOM - 1);
    float dx = xi_sh[0] - xs[(b * NATOM + j) * 3 + 0];
    float dy = xi_sh[1] - xs[(b * NATOM + j) * 3 + 1];
    float dz = xi_sh[2] - xs[(b * NATOM + j) * 3 + 2];
    d_sh[tid] = sqrtf(dx * dx + dy * dy + dz * dz + EPSC);
  }
  __syncthreads();

  const int lane = tid & 31;
  const int wv = tid >> 5;
  const int lm = lane & 15;
  const int lh = lane >> 4;

  // ---- forward: S^l = sum_j softplus(rbf_j @ W1_l + b1_l), via f16 WMMA ----
  for (int it = 0; it < 2; ++it) {
    const int mt = wv * 2 + it;
    H16 a0, a1;
    const float dm = d_sh[mt * 16 + lm];
    const bool zr = (mt * 16 + lm == NATOM - 1);
    build_rbf_frags(a0, a1, dm, zr, lh, gam, mu_sh);
#pragma unroll
    for (int l = 0; l < 3; ++l) {
#pragma unroll
      for (int nt = 0; nt < 4; ++nt) {
        half16 bf0 = *(const half16*)&w1t[l][nt * 16 + lm][lh * 16];
        half16 bf1 = *(const half16*)&w1t[l][nt * 16 + lm][32 + lh * 16];
        F8 acc;
#pragma unroll
        for (int q = 0; q < 8; ++q) acc.e[q] = 0.f;
        acc.v = __builtin_amdgcn_wmma_f32_16x16x32_f16(false, a0.v, false, bf0,
                                                       (short)0, acc.v, false, false);
        acc.v = __builtin_amdgcn_wmma_f32_16x16x32_f16(false, a1.v, false, bf1,
                                                       (short)0, acc.v, false, false);
        const int n = nt * 16 + lm;
        const float bb = b1sh[l][n];
        float part = 0.f;
#pragma unroll
        for (int vg = 0; vg < 8; ++vg) {
          int m = mt * 16 + lh * 8 + vg;     // C layout: row = vg + 8*(lane>=16)
          float z = acc.e[vg] + bb;
          float s = softplusf_(z);
          part += (m == NATOM - 1) ? 0.f : s;
        }
        atomicAdd(&Ssh[l][n], part);
      }
    }
  }
  __syncthreads();

  // ---- per-atom matvecs (tiny): Wsum, h3, readout grad, dE/ds ----
  if (tid < FDIM) {
    const int n = tid;
#pragma unroll
    for (int l = 0; l < 3; ++l) {
      float s = 0.f;
      for (int m = 0; m < FDIM; ++m) s += Ssh[l][m] * W2s[l][m * FDIM + n];
      Aar[l][n] = 1.f + s + 127.f * B2s[l][n];
    }
  }
  __syncthreads();
  if (tid < FDIM) {
    float h0 = Wenc[tid] + benc[tid];
    h3sh[tid] = h0 * Aar[0][tid] * Aar[1][tid] * Aar[2][tid];
  }
  __syncthreads();
  if (tid < FDIM) {
    const int n = tid;
    float u = bt1[n];
    for (int m = 0; m < FDIM; ++m) u += h3sh[m] * Wt1[m * FDIM + n];
    float th = tanhf(u);
    goutsh[n] = Wt2[n] * (1.f - th * th);
  }
  __syncthreads();
  if (tid < FDIM) {
    const int m = tid;
    float g = 0.f;
    for (int n = 0; n < FDIM; ++n) g += Wt1[m * FDIM + n] * goutsh[n];
    float h0 = Wenc[m] + benc[m];
    float gh = g * h0;
    dWsh[0][m] = gh * Aar[1][m] * Aar[2][m];
    dWsh[1][m] = gh * Aar[0][m] * Aar[2][m];
    dWsh[2][m] = gh * Aar[0][m] * Aar[1][m];
  }
  __syncthreads();
  if (tid < FDIM) {
    const int f = tid;
#pragma unroll
    for (int l = 0; l < 3; ++l) {
      float s = 0.f;
      for (int n = 0; n < FDIM; ++n) s += W2s[l][f * FDIM + n] * dWsh[l][n];
      dsvsh[l][f] = s;
    }
  }
  __syncthreads();

  // ---- backward: recompute z, dz = ds*sigmoid(z); drbf = dz @ W1^T (WMMA);
  //      e_j = sum_k drbf_k * rbf_k * (-2*gamma*(d-mu_k)) ----
  for (int it = 0; it < 2; ++it) {
    const int mt = wv * 2 + it;
    H16 a0, a1;
    const float dm = d_sh[mt * 16 + lm];
    const bool zr = (mt * 16 + lm == NATOM - 1);
    build_rbf_frags(a0, a1, dm, zr, lh, gam, mu_sh);
    float eacc[8];
#pragma unroll
    for (int q = 0; q < 8; ++q) eacc[q] = 0.f;

    for (int l = 0; l < 3; ++l) {
#pragma unroll
      for (int nt = 0; nt < 4; ++nt) {
        half16 bf0 = *(const half16*)&w1t[l][nt * 16 + lm][lh * 16];
        half16 bf1 = *(const half16*)&w1t[l][nt * 16 + lm][32 + lh * 16];
        F8 acc;
#pragma unroll
        for (int q = 0; q < 8; ++q) acc.e[q] = 0.f;
        acc.v = __builtin_amdgcn_wmma_f32_16x16x32_f16(false, a0.v, false, bf0,
                                                       (short)0, acc.v, false, false);
        acc.v = __builtin_amdgcn_wmma_f32_16x16x32_f16(false, a1.v, false, bf1,
                                                       (short)0, acc.v, false, false);
        const int n = nt * 16 + lm;
        const float bb = b1sh[l][n];
        const float dsl = dsvsh[l][n];
#pragma unroll
        for (int vg = 0; vg < 8; ++vg) {
          const int ml = lh * 8 + vg;
          const int m = mt * 16 + ml;
          float z = acc.e[vg] + bb;
          float dz = (m == NATOM - 1) ? 0.f : dsl * sigmoidf_(z);
          dzbuf[wv][ml][n] = (_Float16)dz;
        }
      }
      __syncthreads();   // uniform; makes cross-lane dz tile visible

      // dz tile -> A fragments (K = 64 features)
      H16 c0, c1;
      {
        const _Float16* row = &dzbuf[wv][lm][0];
        c0.h[0] = *(const half8v*)(row + lh * 16);
        c0.h[1] = *(const half8v*)(row + 8 + lh * 16);
        c1.h[0] = *(const half8v*)(row + 32 + lh * 16);
        c1.h[1] = *(const half8v*)(row + 40 + lh * 16);
      }
#pragma unroll
      for (int nt = 0; nt < 4; ++nt) {
        half16 bb0 = *(const half16*)&w1o[l][nt * 16 + lm][lh * 16];
        half16 bb1 = *(const half16*)&w1o[l][nt * 16 + lm][32 + lh * 16];
        F8 acc;
#pragma unroll
        for (int q = 0; q < 8; ++q) acc.e[q] = 0.f;
        acc.v = __builtin_amdgcn_wmma_f32_16x16x32_f16(false, c0.v, false, bb0,
                                                       (short)0, acc.v, false, false);
        acc.v = __builtin_amdgcn_wmma_f32_16x16x32_f16(false, c1.v, false, bb1,
                                                       (short)0, acc.v, false, false);
        const int kr = nt * 16 + lm;
        if (kr < KRBF) {
          const float mu = mu_sh[kr];
#pragma unroll
          for (int vg = 0; vg < 8; ++vg) {
            const float dmr = d_sh[mt * 16 + lh * 8 + vg];
            const float t = dmr - mu;
            const float rb = __expf(-gam * t * t);
            eacc[vg] += acc.e[vg] * (-2.f * gam * t * rb);
          }
        }
      }
      __syncthreads();   // uniform; dzbuf reuse for next layer
    }
#pragma unroll
    for (int vg = 0; vg < 8; ++vg)
      atomicAdd(&e_sh[mt * 16 + lh * 8 + vg], eacc[vg]);
  }
  __syncthreads();

  // store e indexed by actual neighbor j (slot 127 hits the unused diagonal)
  {
    int j = (i + 1 + tid) & (NATOM - 1);
    E[(b * NATOM + i) * NATOM + j] = e_sh[tid];
  }
}

// Kernel 2: deterministic force assembly + mean centering.
// grad_i = sum_{j!=i} (e_ij + e_ji) * (x_i - x_j) / d_ij ; out = -(grad - mean).
__global__ __launch_bounds__(128) void schnet_assemble(
    const float* __restrict__ xs, const float* __restrict__ E,
    float* __restrict__ out)
{
  __shared__ float xsh[NATOM * 3];
  __shared__ float red[3][NATOM];
  const int b = blockIdx.x;
  const int i = threadIdx.x;
  for (int t = i; t < NATOM * 3; t += 128) xsh[t] = xs[b * NATOM * 3 + t];
  __syncthreads();
  const float xi0 = xsh[i * 3 + 0];
  const float xi1 = xsh[i * 3 + 1];
  const float xi2 = xsh[i * 3 + 2];
  const float* Eb = E + (size_t)b * NATOM * NATOM;
  float f0 = 0.f, f1 = 0.f, f2 = 0.f;
  for (int j = 0; j < NATOM; ++j) {
    if (j == i) continue;
    float dx = xi0 - xsh[j * 3 + 0];
    float dy = xi1 - xsh[j * 3 + 1];
    float dz = xi2 - xsh[j * 3 + 2];
    float d = sqrtf(dx * dx + dy * dy + dz * dz + EPSC);
    float w = (Eb[i * NATOM + j] + Eb[j * NATOM + i]) / d;
    f0 += w * dx; f1 += w * dy; f2 += w * dz;
  }
  red[0][i] = f0; red[1][i] = f1; red[2][i] = f2;
  __syncthreads();
  for (int s = NATOM / 2; s > 0; s >>= 1) {
    if (i < s) {
      red[0][i] += red[0][i + s];
      red[1][i] += red[1][i + s];
      red[2][i] += red[2][i + s];
    }
    __syncthreads();
  }
  const float m0 = red[0][0] * (1.f / NATOM);
  const float m1 = red[1][0] * (1.f / NATOM);
  const float m2 = red[2][0] * (1.f / NATOM);
  out[b * NATOM * 3 + i * 3 + 0] = -(f0 - m0);
  out[b * NATOM * 3 + i * 3 + 1] = -(f1 - m1);
  out[b * NATOM * 3 + i * 3 + 2] = -(f2 - m2);
}

extern "C" void kernel_launch(void* const* d_in, const int* in_sizes, int n_in,
                              void* d_out, int out_size, void* d_ws, size_t ws_size,
                              hipStream_t stream) {
  (void)in_sizes; (void)n_in; (void)out_size; (void)ws_size;
  const float* xs    = (const float*)d_in[1];
  const float* mus   = (const float*)d_in[2];
  const float* gamma = (const float*)d_in[3];
  const float* Wenc  = (const float*)d_in[4];
  const float* benc  = (const float*)d_in[5];
  const float* w1_0  = (const float*)d_in[6];
  const float* b1_0  = (const float*)d_in[7];
  const float* w2_0  = (const float*)d_in[8];
  const float* b2_0  = (const float*)d_in[9];
  const float* w1_1  = (const float*)d_in[10];
  const float* b1_1  = (const float*)d_in[11];
  const float* w2_1  = (const float*)d_in[12];
  const float* b2_1  = (const float*)d_in[13];
  const float* w1_2  = (const float*)d_in[14];
  const float* b1_2  = (const float*)d_in[15];
  const float* w2_2  = (const float*)d_in[16];
  const float* b2_2  = (const float*)d_in[17];
  const float* Wt1   = (const float*)d_in[18];
  const float* bt1   = (const float*)d_in[19];
  const float* Wt2   = (const float*)d_in[20];
  float* E = (float*)d_ws;  // B*N*N floats = 2 MB scratch

  schnet_pair_grad<<<dim3(NBATCH * NATOM), dim3(128), 0, stream>>>(
      xs, mus, gamma, Wenc, benc,
      w1_0, b1_0, w2_0, b2_0,
      w1_1, b1_1, w2_1, b2_1,
      w1_2, b1_2, w2_2, b2_2,
      Wt1, bt1, Wt2, E);
  schnet_assemble<<<dim3(NBATCH), dim3(NATOM), 0, stream>>>(xs, E, (float*)d_out);
}
